// My_Attention_90829968376419
// MI455X (gfx1250) — matrix operand order repaired and checked by reading, not compiled, not run on hardware
//
#include <hip/hip_runtime.h>
#include <hip/hip_bf16.h>
#include <stdint.h>

// ---------------------------------------------------------------------------
// Linformer attention, MI455X (gfx1250), wave32.
//  - V_WMMA_F32_16X16X32_BF16 for all matrix math (f32 accum)
//  - GLOBAL_LOAD_ASYNC_TO_LDS_B128 + double-buffered LDS + s_wait_asynccnt
//    for GEMM tile staging (CDNA5 async path, overlaps copy with WMMA)
// Pipeline: cvt fp32->bf16 ; QKV GEMM ; kp/vp GEMM (+be) ; fused attn ;
//           output GEMM (+bo, fp32 out).
// ---------------------------------------------------------------------------

typedef __attribute__((ext_vector_type(16))) __bf16 v16bf;
typedef __attribute__((ext_vector_type(8)))  __bf16 bf16x8;
typedef __attribute__((ext_vector_type(8)))  float  v8f;

__device__ __forceinline__ __bf16 f2bf(float f) {
  union { float f; unsigned u; } x; x.f = f;
  unsigned r = x.u + 0x7fffu + ((x.u >> 16) & 1u);   // round-to-nearest-even
  unsigned short h = (unsigned short)(r >> 16);
  return __builtin_bit_cast(__bf16, h);
}

// low 32 bits of a generic pointer to __shared__ == workgroup-relative LDS offset
__device__ __forceinline__ unsigned lds_off_of(const void* p) {
  return (unsigned)(uintptr_t)p;
}

// Async DMA: 16 bytes global -> LDS per lane.  Tracked by ASYNCcnt.
__device__ __forceinline__ void async_b128(unsigned lds_off, const void* gptr) {
  asm volatile("global_load_async_to_lds_b128 %0, %1, off"
               :: "v"(lds_off), "v"((unsigned long long)(uintptr_t)gptr)
               : "memory");
}
__device__ __forceinline__ void wait_async0() {
  asm volatile("s_wait_asynccnt 0" ::: "memory");
}

// A-fragment (16x32 bf16, MxK): lane holds K = {h*8..h*8+7, 16+h*8..16+h*8+7}
__device__ __forceinline__ v16bf load_frag_a(const __bf16* row, int half) {
  bf16x8 lo = *(const bf16x8*)(row + half * 8);
  bf16x8 hi = *(const bf16x8*)(row + 16 + half * 8);
  v16bf r;
#pragma unroll
  for (int i = 0; i < 8; ++i) { r[i] = lo[i]; r[i + 8] = hi[i]; }
  return r;
}

// B-fragment (32x16 bf16, KxN): lane (col = lane%16) holds K = {h*8..h*8+7, 16+h*8..16+h*8+7}
__device__ __forceinline__ v16bf load_frag_b(const __bf16* col, int half) {
  bf16x8 lo = *(const bf16x8*)(col + half * 8);
  bf16x8 hi = *(const bf16x8*)(col + 16 + half * 8);
  v16bf r;
#pragma unroll
  for (int i = 0; i < 8; ++i) { r[i] = lo[i]; r[i + 8] = hi[i]; }
  return r;
}

// ---------------------------------------------------------------------------
// fp32 -> bf16 elementwise
// ---------------------------------------------------------------------------
__global__ void lin_f32_to_bf16(const float* __restrict__ in,
                                __bf16* __restrict__ out, long long n) {
  long long i = (long long)blockIdx.x * blockDim.x + threadIdx.x;
  long long stride = (long long)gridDim.x * blockDim.x;
  for (; i < n; i += stride) out[i] = f2bf(in[i]);
}

// ---------------------------------------------------------------------------
// bf16 WMMA GEMM, async double-buffered.  C[m,n] = sum_k A[m,k] * B'[k,n]
//   BT=1: B stored [N,K] row-major (C = A·B^T), fully async staging
//   BT=0: B stored [K,N] row-major, scalar-transpose staging for B
// BM=BN=64, BK=64.  128 threads (4 waves), 2x2 wave grid, 2x2 16x16 per wave,
// 8 WMMA per K-iteration.  biasN per col / biasM per row (within batch).
// ---------------------------------------------------------------------------
#define LGB_BK 64
#define LGB_LD 72   // row stride (elems), 144B: 16B-aligned, bank-staggered

template <int BT, int CF32>
__global__ __launch_bounds__(128)
void lin_gemm_bf16_wmma(const __bf16* __restrict__ A, const __bf16* __restrict__ B,
                        void* __restrict__ Cv,
                        int K, int lda, int ldb, int ldc,
                        long long sA, long long sB, long long sC,
                        const float* __restrict__ biasN,
                        const float* __restrict__ biasM) {
  __shared__ __bf16 As[2][64][LGB_LD];   // row m contiguous in k
  __shared__ __bf16 Bs[2][64][LGB_LD];   // col n contiguous in k

  const int tid  = threadIdx.x;
  const int lane = tid & 31, w = tid >> 5;
  const int l16  = lane & 15, half = lane >> 4;
  const int wm   = w >> 1,   wn = w & 1;
  const int n0   = blockIdx.x * 64;
  const int m0   = blockIdx.y * 64;
  const long long bz = blockIdx.z;

  const __bf16* Ab = A + sA * bz;
  const __bf16* Bb = B + sB * bz;

  // per-thread chunk coords: 512 x 16B chunks per 64x64 tile, 4 per thread
  const unsigned asBase = lds_off_of(&As[0][0][0]);
  const unsigned bsBase = lds_off_of(&Bs[0][0][0]);
  const unsigned bufStride = 64u * LGB_LD * 2u;

  auto issueA = [&](int buf, int kt) {
#pragma unroll
    for (int j = 0; j < 4; ++j) {
      int id = tid + j * 128;
      int row = id >> 2, col = (id & 3) * 16;
      async_b128(asBase + (unsigned)buf * bufStride + (unsigned)(row * LGB_LD + col) * 2u,
                 Ab + (long long)(m0 + row) * lda + kt + col);
    }
  };
  auto issueB = [&](int buf, int kt) {
    if (BT) {
#pragma unroll
      for (int j = 0; j < 4; ++j) {
        int id = tid + j * 128;
        int row = id >> 2, col = (id & 3) * 16;
        async_b128(bsBase + (unsigned)buf * bufStride + (unsigned)(row * LGB_LD + col) * 2u,
                   Bb + (long long)(n0 + row) * ldb + kt + col);
      }
    } else {  // transpose 64x64 [k][n] -> Bs[n][k], scalar
#pragma unroll
      for (int i = 0; i < 32; ++i) {
        int idx = tid + i * 128;
        int kr = idx >> 6, nc = idx & 63;
        Bs[buf][nc][kr] = Bb[(long long)(kt + kr) * ldb + n0 + nc];
      }
    }
  };

  v8f zero = {0.f,0.f,0.f,0.f,0.f,0.f,0.f,0.f};
  v8f acc[2][2];
#pragma unroll
  for (int i = 0; i < 2; ++i)
#pragma unroll
    for (int j = 0; j < 2; ++j) acc[i][j] = zero;

  const int nk = K / LGB_BK;
  issueA(0, 0);
  issueB(0, 0);

  for (int it = 0; it < nk; ++it) {
    const int cur = it & 1;
    if (it + 1 < nk) {            // prefetch next tile into other buffer
      issueA(cur ^ 1, (it + 1) * LGB_BK);
      issueB(cur ^ 1, (it + 1) * LGB_BK);
      // retire current tile's asyncs (in-order completion), keep next in flight
      if (BT) asm volatile("s_wait_asynccnt 8" ::: "memory");
      else    asm volatile("s_wait_asynccnt 4" ::: "memory");
    } else {
      wait_async0();
    }
    __syncthreads();

#pragma unroll
    for (int ks = 0; ks < 2; ++ks) {
      v16bf af[2], bfg[2];
#pragma unroll
      for (int ms = 0; ms < 2; ++ms)
        af[ms] = load_frag_a(&As[cur][wm * 32 + ms * 16 + l16][ks * 32], half);
#pragma unroll
      for (int ns = 0; ns < 2; ++ns)
        bfg[ns] = load_frag_b(&Bs[cur][wn * 32 + ns * 16 + l16][ks * 32], half);
#pragma unroll
      for (int ms = 0; ms < 2; ++ms)
#pragma unroll
        for (int ns = 0; ns < 2; ++ns)
          acc[ms][ns] = __builtin_amdgcn_wmma_f32_16x16x32_bf16(
              false, af[ms], false, bfg[ns], (short)0, acc[ms][ns], false, false);
    }
    __syncthreads();              // all waves done reading before overwrite
  }

  float*  Cf = (float*)Cv;
  __bf16* Ch = (__bf16*)Cv;
#pragma unroll
  for (int ms = 0; ms < 2; ++ms)
#pragma unroll
    for (int ns = 0; ns < 2; ++ns)
#pragma unroll
      for (int r = 0; r < 8; ++r) {
        int mrow = m0 + wm * 32 + ms * 16 + r + 8 * half;  // C layout: m = r + 8*half
        int ncol = n0 + wn * 32 + ns * 16 + l16;           //           n = lane%16
        float v = acc[ms][ns][r];
        if (biasN) v += biasN[ncol];
        if (biasM) v += biasM[mrow];
        long long ci = sC * bz + (long long)mrow * ldc + ncol;
        if (CF32) Cf[ci] = v; else Ch[ci] = f2bf(v);
      }
}

// ---------------------------------------------------------------------------
// Fused Linformer attention: per (b, h, 64 query rows).
//   dots = q @ kp^T * 0.125 ; softmax over P=256 ; out = attn @ vp
// 4 waves, one 16-row q tile per wave. kp staged via async-to-LDS.
// ---------------------------------------------------------------------------
__global__ __launch_bounds__(128)
void lin_attn_wmma(const __bf16* __restrict__ qkv,  // [B*4096, 3072], q at +0
                   const __bf16* __restrict__ kp,   // [B, 256, 1024]
                   const __bf16* __restrict__ vp,   // [B, 256, 1024]
                   __bf16* __restrict__ outp) {     // [B*4096, 1024]
  __shared__ __bf16 kps[256][72];        // [p][d] : B-frag for dots (K=d)
  __shared__ __bf16 vps[64][264];        // [d][p] : B-frag for out  (K=p)
  __shared__ __bf16 attn_s[4][16][264];  // per wave: [row][p] A-frag

  const int tid  = threadIdx.x;
  const int lane = tid & 31, w = tid >> 5;
  const int l16  = lane & 15, half = lane >> 4;
  const int n0 = blockIdx.x * 64;
  const int h  = blockIdx.y;
  const int b  = blockIdx.z;

  {  // kp -> kps[p][d] via async DMA (2048 x 16B chunks, 16 per thread)
    const __bf16* src = kp + (long long)b * 256 * 1024 + h * 64;
    const unsigned base = lds_off_of(&kps[0][0]);
#pragma unroll
    for (int i = 0; i < 16; ++i) {
      int chunk = tid + i * 128;
      int p = chunk >> 3, d8 = (chunk & 7) * 8;
      async_b128(base + (unsigned)(p * 72 + d8) * 2u, src + (long long)p * 1024 + d8);
    }
  }
  {  // vp -> vps[d][p] (transposed), scalar
    const __bf16* src = vp + (long long)b * 256 * 1024 + h * 64;
    for (int i = 0; i < 128; ++i) {
      int idx = tid + i * 128;
      int d = idx & 63, p = idx >> 6;
      vps[d][p] = src[(long long)p * 1024 + d];
    }
  }
  wait_async0();
  __syncthreads();

  // q fragments for this wave's 16 rows (K = DH = 64 -> 2 k-steps)
  const __bf16* qrow = qkv + (long long)(b * 4096 + n0 + w * 16 + l16) * 3072 + h * 64;
  v16bf qf[2];
#pragma unroll
  for (int ks = 0; ks < 2; ++ks) qf[ks] = load_frag_a(qrow + ks * 32, half);

  // dots: 16 j-tiles of 16 cols each (P = 256)
  v8f zero = {0.f,0.f,0.f,0.f,0.f,0.f,0.f,0.f};
  v8f dacc[16];
#pragma unroll
  for (int jt = 0; jt < 16; ++jt) {
    v8f c = zero;
#pragma unroll
    for (int ks = 0; ks < 2; ++ks) {
      v16bf bfg = load_frag_b(&kps[jt * 16 + l16][ks * 32], half);
      c = __builtin_amdgcn_wmma_f32_16x16x32_bf16(false, qf[ks], false, bfg,
                                                  (short)0, c, false, false);
    }
    dacc[jt] = c;
  }

  // softmax: row m = r + 8*half lives across lanes (half*16 .. half*16+15)
  const float SCALE = 0.125f;  // 64^-0.5
#pragma unroll
  for (int r = 0; r < 8; ++r) {
    float m = -3.0e38f;
#pragma unroll
    for (int jt = 0; jt < 16; ++jt) {
      float s = dacc[jt][r] * SCALE; dacc[jt][r] = s; m = fmaxf(m, s);
    }
#pragma unroll
    for (int off = 1; off < 16; off <<= 1) m = fmaxf(m, __shfl_xor(m, off, 16));
    float sum = 0.f;
#pragma unroll
    for (int jt = 0; jt < 16; ++jt) {
      float e = __expf(dacc[jt][r] - m); dacc[jt][r] = e; sum += e;
    }
#pragma unroll
    for (int off = 1; off < 16; off <<= 1) sum += __shfl_xor(sum, off, 16);
    float inv = 1.0f / sum;
#pragma unroll
    for (int jt = 0; jt < 16; ++jt)
      attn_s[w][r + 8 * half][jt * 16 + l16] = f2bf(dacc[jt][r] * inv);
  }

  // out = attn @ vp : K = 256 -> 8 k-steps ; DH = 64 -> 4 col tiles
  v16bf afr[8];
#pragma unroll
  for (int ks = 0; ks < 8; ++ks) afr[ks] = load_frag_a(&attn_s[w][l16][ks * 32], half);

#pragma unroll
  for (int dt = 0; dt < 4; ++dt) {
    v8f oc = zero;
#pragma unroll
    for (int ks = 0; ks < 8; ++ks) {
      v16bf bfg = load_frag_b(&vps[dt * 16 + l16][ks * 32], half);
      oc = __builtin_amdgcn_wmma_f32_16x16x32_bf16(false, afr[ks], false, bfg,
                                                   (short)0, oc, false, false);
    }
#pragma unroll
    for (int r = 0; r < 8; ++r) {
      int row = n0 + w * 16 + r + 8 * half;
      outp[(long long)(b * 4096 + row) * 1024 + h * 64 + dt * 16 + l16] = f2bf(oc[r]);
    }
  }
}

// ---------------------------------------------------------------------------
extern "C" void kernel_launch(void* const* d_in, const int* in_sizes, int n_in,
                              void* d_out, int out_size, void* d_ws, size_t ws_size,
                              hipStream_t stream) {
  (void)in_sizes; (void)n_in; (void)out_size; (void)ws_size;
  const float* x    = (const float*)d_in[0];
  const float* Wqkv = (const float*)d_in[1];
  const float* We   = (const float*)d_in[2];
  const float* be   = (const float*)d_in[3];
  const float* Wo   = (const float*)d_in[4];
  const float* bo   = (const float*)d_in[5];
  float* out = (float*)d_out;

  const long long Bq = 8, Nq = 4096, DIM = 1024, Hh = 16, Pp = 256;
  const long long BN_ = Bq * Nq;  // 32768

  char* ws = (char*)d_ws;
  size_t off = 0;
  auto alloc = [&](size_t bytes) {
    char* p = ws + off; off += (bytes + 255) & ~(size_t)255; return p;
  };
  __bf16* x_bf    = (__bf16*)alloc((size_t)BN_ * DIM * 2);
  __bf16* Wqkv_bf = (__bf16*)alloc((size_t)3 * DIM * DIM * 2);
  __bf16* We_bf   = (__bf16*)alloc((size_t)Pp * Nq * 2);
  __bf16* Wo_bf   = (__bf16*)alloc((size_t)DIM * DIM * 2);
  __bf16* qkv_bf  = (__bf16*)alloc((size_t)BN_ * 3 * DIM * 2);
  __bf16* kp_bf   = (__bf16*)alloc((size_t)Bq * Pp * DIM * 2);
  __bf16* vp_bf   = (__bf16*)alloc((size_t)Bq * Pp * DIM * 2);
  __bf16* ao_bf   = (__bf16*)alloc((size_t)BN_ * DIM * 2);

  lin_f32_to_bf16<<<2048, 256, 0, stream>>>(x,    x_bf,    BN_ * DIM);
  lin_f32_to_bf16<<<1024, 256, 0, stream>>>(Wqkv, Wqkv_bf, 3 * DIM * DIM);
  lin_f32_to_bf16<<<1024, 256, 0, stream>>>(We,   We_bf,   Pp * Nq);
  lin_f32_to_bf16<<<1024, 256, 0, stream>>>(Wo,   Wo_bf,   DIM * DIM);

  // qkv = x @ Wqkv^T    [32768 x 3072], K = 1024
  lin_gemm_bf16_wmma<1, 0><<<dim3(48, 512, 1), 128, 0, stream>>>(
      x_bf, Wqkv_bf, qkv_bf, (int)DIM,
      (int)DIM, (int)DIM, (int)(3 * DIM), 0LL, 0LL, 0LL, nullptr, nullptr);

  // kp[b] = We @ K_b + be   [256 x 1024], K = 4096, batched over b
  lin_gemm_bf16_wmma<0, 0><<<dim3(16, 4, 8), 128, 0, stream>>>(
      We_bf, qkv_bf + DIM, kp_bf, (int)Nq,
      (int)Nq, (int)(3 * DIM), (int)DIM, 0LL, Nq * 3 * DIM, Pp * DIM, nullptr, be);
  // vp[b] = We @ V_b + be
  lin_gemm_bf16_wmma<0, 0><<<dim3(16, 4, 8), 128, 0, stream>>>(
      We_bf, qkv_bf + 2 * DIM, vp_bf, (int)Nq,
      (int)Nq, (int)(3 * DIM), (int)DIM, 0LL, Nq * 3 * DIM, Pp * DIM, nullptr, be);

  // fused softmax(q·kp^T·scale)·vp
  lin_attn_wmma<<<dim3(64, (unsigned)Hh, (unsigned)Bq), 128, 0, stream>>>(
      qkv_bf, kp_bf, vp_bf, ao_bf);

  // out = ao @ Wo^T + bo   [32768 x 1024], fp32 output
  lin_gemm_bf16_wmma<1, 1><<<dim3(16, 512, 1), 128, 0, stream>>>(
      ao_bf, Wo_bf, out, (int)DIM,
      (int)DIM, (int)DIM, (int)DIM, 0LL, 0LL, 0LL, bo, nullptr);
}
